// EncoderBasic_4879082848584
// MI455X (gfx1250) — matrix operand order repaired and checked by reading, not compiled, not run on hardware
//
#include <hip/hip_runtime.h>
#include <hip/hip_bf16.h>
#include <stdint.h>

#define T_SEQ 4096
#define H     1024
#define H3    3072
#define NLAYER 2
#define REC_BLOCKS 16
#define REC_THREADS 64

typedef __attribute__((ext_vector_type(16))) __bf16 v16bf;
typedef __attribute__((ext_vector_type(8)))  float  v8f;

__device__ __forceinline__ uint16_t f32_to_bf16_rne(float f) {
  uint32_t u = __float_as_uint(f);
  uint32_t r = u + 0x7fffu + ((u >> 16) & 1u);
  return (uint16_t)(r >> 16);
}

// ---- CDNA5 async global->LDS (ASYNCcnt) helpers ----
__device__ __forceinline__ void async_copy_b128(uint32_t lds_addr, const void* gptr) {
  asm volatile("global_load_async_to_lds_b128 %0, %1, off"
               :: "v"(lds_addr), "v"((uint64_t)(uintptr_t)gptr)
               : "memory");
}
__device__ __forceinline__ void wait_async_le4() {
  asm volatile("s_wait_asynccnt 0x4" ::: "memory");
}
__device__ __forceinline__ void wait_async_0() {
  asm volatile("s_wait_asynccnt 0x0" ::: "memory");
}

// ---------------- embedding gather -> bf16 ----------------
__global__ void k_embed(const int* __restrict__ idx, const float* __restrict__ emb,
                        uint16_t* __restrict__ xb) {
  int t = blockIdx.x;
  int row = idx[t];
  const float* src = emb + (size_t)row * H;
  uint16_t* dst = xb + (size_t)t * H;
  for (int i = threadIdx.x; i < H; i += blockDim.x)
    dst[i] = f32_to_bf16_rne(src[i]);
}

// ---------------- f32 -> bf16 elementwise (W_ih) ----------------
__global__ void k_f32_to_bf16(const float* __restrict__ in, uint16_t* __restrict__ out, int n) {
  int i = blockIdx.x * blockDim.x + threadIdx.x;
  if (i < n) out[i] = f32_to_bf16_rne(in[i]);
}

// ---------------- W_hh -> transposed packed bf16 pairs ----------------
__global__ void k_pack_whh(const float* __restrict__ whh, uint32_t* __restrict__ out) {
  int i = blockIdx.x * blockDim.x + threadIdx.x;
  if (i >= NLAYER * 3 * (H / 2) * H) return;
  int j  = i & 1023;
  int k2 = (i >> 10) & 511;
  int lg = i >> 19;  // l*3+g
  const float* rowp = whh + ((size_t)(lg * 1024 + j)) * H + 2 * k2;
  uint32_t lo = f32_to_bf16_rne(rowp[0]);
  uint32_t hi = f32_to_bf16_rne(rowp[1]);
  out[i] = lo | (hi << 16);
}

// ---------------- gi = A @ B^T + bias : LDS-staged, async double-buffered WMMA ----------------
// A: [T][H] bf16 row-major; B(W_ih): [3H][H] bf16 row-major; gi: [T][3H] f32.
// Block = 256 thr (8 waves, 2x4), macro-tile 128x128, K-step 32.
#define LROW 80            // LDS row stride in bytes (32 bf16 = 64B data + 16B pad)
#define PANEL (128 * LROW) // 10240 B per operand panel
#define BUFSZ (2 * PANEL)  // A + B panels per buffer
__global__ void __launch_bounds__(256) k_gemm_gi(const uint16_t* __restrict__ A,
                                                 const uint16_t* __restrict__ B,
                                                 const float* __restrict__ bias,
                                                 float* __restrict__ gi) {
  __shared__ __align__(16) char smem[2 * BUFSZ];  // 40 KB, double buffered

  const int lane = threadIdx.x & 31;
  const int wave = threadIdx.x >> 5;
  const int wm = wave >> 2;        // 0..1  (M offset wm*64)
  const int wn = wave & 3;         // 0..3  (N offset wn*32)
  const int hi = lane >> 4;        // lane half
  const int n16 = lane & 15;
  const int bm = blockIdx.y;       // 0..31
  const int bn = blockIdx.x;       // 0..23

  // staging role: thread t copies 32B of row (t>>1), half (t&1), for both panels
  const int srow  = threadIdx.x >> 1;
  const int shalf = threadIdx.x & 1;
  const uint16_t* gA = A + (size_t)(bm * 128 + srow) * H + shalf * 16;
  const uint16_t* gB = B + (size_t)(bn * 128 + srow) * H + shalf * 16;
  const uint32_t lbase = (uint32_t)(uintptr_t)smem;  // flat LDS addr[31:0] == LDS offset
  const uint32_t ldsA = lbase + srow * LROW + shalf * 32;
  const uint32_t ldsB = lbase + PANEL + srow * LROW + shalf * 32;

  auto issue = [&](int kt) {
    const uint32_t boff = (kt & 1) ? (uint32_t)BUFSZ : 0u;
    const uint16_t* pa = gA + kt * 32;
    const uint16_t* pb = gB + kt * 32;
    async_copy_b128(ldsA + boff,      pa);
    async_copy_b128(ldsA + boff + 16, pa + 8);
    async_copy_b128(ldsB + boff,      pb);
    async_copy_b128(ldsB + boff + 16, pb + 8);
  };

  v8f c[4][2] = {};
  union frag { uint4 q[2]; v16bf v; };

  issue(0);
  const int NK = H / 32;
  for (int kt = 0; kt < NK; ++kt) {
    if (kt + 1 < NK) { issue(kt + 1); wait_async_le4(); }
    else             { wait_async_0(); }
    __syncthreads();  // all waves' portions of buf[kt&1] are resident

    const char* bufA = smem + ((kt & 1) ? BUFSZ : 0);
    const char* bufB = bufA + PANEL;

    frag a[4], b;
#pragma unroll
    for (int sm = 0; sm < 4; ++sm) {
      const char* p = bufA + (wm * 64 + sm * 16 + n16) * LROW;
      a[sm].q[0] = *(const uint4*)(p + hi * 16);        // K = hi*8 .. hi*8+7
      a[sm].q[1] = *(const uint4*)(p + 32 + hi * 16);   // K = 16+hi*8 ..
    }
#pragma unroll
    for (int sn = 0; sn < 2; ++sn) {
      const char* p = bufB + (wn * 32 + sn * 16 + n16) * LROW;
      b.q[0] = *(const uint4*)(p + hi * 16);            // K = hi*8 .. hi*8+7
      b.q[1] = *(const uint4*)(p + 32 + hi * 16);       // K = 16+hi*8 ..
#pragma unroll
      for (int sm = 0; sm < 4; ++sm)
        c[sm][sn] = __builtin_amdgcn_wmma_f32_16x16x32_bf16(
            false, a[sm].v, false, b.v, (short)0, c[sm][sn], false, false);
    }
    __syncthreads();  // done reading buf[kt&1] before it is refilled at kt+2
  }

#pragma unroll
  for (int sn = 0; sn < 2; ++sn) {
    int col = bn * 128 + wn * 32 + sn * 16 + n16;
    float bv = bias[col];
#pragma unroll
    for (int sm = 0; sm < 4; ++sm) {
#pragma unroll
      for (int r = 0; r < 8; ++r) {
        int row = bm * 128 + wm * 64 + sm * 16 + (hi ? 8 : 0) + r;
        gi[(size_t)row * H3 + col] = c[sm][sn][r] + bv;
      }
    }
  }
}

// ---------------- per-call init (graph-replay safe) ----------------
__global__ void k_init_rec(float* __restrict__ h, uint32_t* __restrict__ ctrl) {
  int i = threadIdx.x + blockIdx.x * blockDim.x;
  if (i < H) h[i] = 0.f;
  if (i < 2) ctrl[i] = 0u;
}

// ---------------- persistent-grid barrier (generation counting) ----------------
__device__ __forceinline__ void grid_barrier(uint32_t* cnt, uint32_t* gen) {
  __syncthreads();
  if (threadIdx.x == 0) {
    uint32_t g = __hip_atomic_load(gen, __ATOMIC_RELAXED, __HIP_MEMORY_SCOPE_AGENT);
    uint32_t a = __hip_atomic_fetch_add(cnt, 1u, __ATOMIC_ACQ_REL, __HIP_MEMORY_SCOPE_AGENT);
    if (a == REC_BLOCKS - 1) {
      __hip_atomic_store(cnt, 0u, __ATOMIC_RELAXED, __HIP_MEMORY_SCOPE_AGENT);
      __hip_atomic_fetch_add(gen, 1u, __ATOMIC_RELEASE, __HIP_MEMORY_SCOPE_AGENT);
    } else {
      while (__hip_atomic_load(gen, __ATOMIC_ACQUIRE, __HIP_MEMORY_SCOPE_AGENT) == g)
        __builtin_amdgcn_s_sleep(2);
    }
  }
  __syncthreads();
}

// ---------------- sequential GRU scan: 1 thread owns hidden unit j ----------------
__global__ void k_gru_rec(const float* __restrict__ gi,      // [T][3H]
                          const uint32_t* __restrict__ wpk,  // [3][512][1024] packed bf16 pairs
                          const float* __restrict__ bhh,     // [3H]
                          float* __restrict__ hbuf,          // [H]
                          uint32_t* __restrict__ ctrl,       // {cnt, gen}
                          float* __restrict__ y_f32,         // [T][H] or null
                          uint16_t* __restrict__ y_bf16,     // [T][H] or null
                          float* __restrict__ h_final) {     // [H]
  __shared__ float hs[H];
  const int j = blockIdx.x * REC_THREADS + threadIdx.x;
  const float br = bhh[j], bz = bhh[H + j], bn = bhh[2 * H + j];
  const uint32_t* wr = wpk + j;
  const uint32_t* wz = wpk + 512 * 1024 + j;
  const uint32_t* wn = wpk + 2 * 512 * 1024 + j;
  float last = 0.f;

  for (int t = 0; t < T_SEQ; ++t) {
#pragma unroll
    for (int i = 0; i < H / REC_THREADS; ++i) {
      int i0 = threadIdx.x + i * REC_THREADS;
      hs[i0] = __hip_atomic_load(hbuf + i0, __ATOMIC_RELAXED, __HIP_MEMORY_SCOPE_AGENT);
    }
    __syncthreads();

    float ar = 0.f, az = 0.f, an = 0.f;
#pragma unroll 4
    for (int k2 = 0; k2 < H / 2; ++k2) {
      float h0 = hs[2 * k2];
      float h1 = hs[2 * k2 + 1];
      uint32_t pr = wr[k2 << 10];
      uint32_t pz = wz[k2 << 10];
      uint32_t pn = wn[k2 << 10];
      ar = fmaf(h0, __uint_as_float(pr << 16), ar);
      ar = fmaf(h1, __uint_as_float(pr & 0xffff0000u), ar);
      az = fmaf(h0, __uint_as_float(pz << 16), az);
      az = fmaf(h1, __uint_as_float(pz & 0xffff0000u), az);
      an = fmaf(h0, __uint_as_float(pn << 16), an);
      an = fmaf(h1, __uint_as_float(pn & 0xffff0000u), an);
    }

    const float* git = gi + (size_t)t * H3;
    float r = 1.f / (1.f + expf(-(git[j] + ar + br)));
    float z = 1.f / (1.f + expf(-(git[H + j] + az + bz)));
    float n = tanhf(git[2 * H + j] + r * (an + bn));
    float hn = (1.f - z) * n + z * hs[j];
    last = hn;

    __hip_atomic_store(hbuf + j, hn, __ATOMIC_RELAXED, __HIP_MEMORY_SCOPE_AGENT);
    if (y_f32)  y_f32[(size_t)t * H + j] = hn;
    if (y_bf16) y_bf16[(size_t)t * H + j] = f32_to_bf16_rne(hn);
    grid_barrier(ctrl, ctrl + 1);
  }
  h_final[j] = last;
}

extern "C" void kernel_launch(void* const* d_in, const int* in_sizes, int n_in,
                              void* d_out, int out_size, void* d_ws, size_t ws_size,
                              hipStream_t stream) {
  const int*   idx = (const int*)d_in[0];
  const float* emb = (const float*)d_in[1];
  const float* wih = (const float*)d_in[2];
  const float* whh = (const float*)d_in[3];
  const float* bih = (const float*)d_in[4];
  const float* bhh = (const float*)d_in[5];
  float* out = (float*)d_out;
  (void)in_sizes; (void)n_in; (void)out_size; (void)ws_size;

  char* ws = (char*)d_ws;
  size_t off = 0;
  auto alloc = [&](size_t bytes) -> void* {
    off = (off + 255) & ~(size_t)255;
    void* p = ws + off;
    off += bytes;
    return p;
  };
  uint16_t* x_bf   = (uint16_t*)alloc((size_t)T_SEQ * H * 2);
  uint16_t* y0_bf  = (uint16_t*)alloc((size_t)T_SEQ * H * 2);
  uint16_t* wih_bf = (uint16_t*)alloc((size_t)NLAYER * H3 * H * 2);
  uint32_t* whh_pk = (uint32_t*)alloc((size_t)NLAYER * 3 * (H / 2) * H * 4);
  float*    gi     = (float*)alloc((size_t)T_SEQ * H3 * 4);
  float*    hbuf   = (float*)alloc((size_t)H * 4);
  uint32_t* ctrl   = (uint32_t*)alloc(64);

  k_embed<<<T_SEQ, 256, 0, stream>>>(idx, emb, x_bf);
  int nwih = NLAYER * H3 * H;
  k_f32_to_bf16<<<(nwih + 255) / 256, 256, 0, stream>>>(wih, wih_bf, nwih);
  int npk = NLAYER * 3 * (H / 2) * H;
  k_pack_whh<<<(npk + 255) / 256, 256, 0, stream>>>(whh, whh_pk);

  dim3 ggrid(H3 / 128, T_SEQ / 128);  // (24, 32)

  // ---- layer 0 ----
  k_gemm_gi<<<ggrid, 256, 0, stream>>>(x_bf, wih_bf, bih, gi);
  k_init_rec<<<4, 256, 0, stream>>>(hbuf, ctrl);
  k_gru_rec<<<REC_BLOCKS, REC_THREADS, 0, stream>>>(
      gi, whh_pk, bhh, hbuf, ctrl,
      /*y_f32=*/nullptr, /*y_bf16=*/y0_bf, out + (size_t)T_SEQ * H);

  // ---- layer 1 ----
  k_gemm_gi<<<ggrid, 256, 0, stream>>>(y0_bf, wih_bf + (size_t)H3 * H, bih + H3, gi);
  k_init_rec<<<4, 256, 0, stream>>>(hbuf, ctrl);
  k_gru_rec<<<REC_BLOCKS, REC_THREADS, 0, stream>>>(
      gi, whh_pk + (size_t)3 * (H / 2) * H, bhh + H3, hbuf, ctrl,
      /*y_f32=*/out, /*y_bf16=*/nullptr, out + (size_t)T_SEQ * H + H);
}
